// DynamicResASPP_29686813950115
// MI455X (gfx1250) — matrix operand order repaired and checked by reading, not compiled, not run on hardware
//
#include <hip/hip_runtime.h>
#include <hip/hip_bf16.h>
#include <math.h>

typedef __attribute__((ext_vector_type(8)))  float  v8f;
typedef __attribute__((ext_vector_type(16))) __bf16 v16bf;

#define HW 128
#define BROWS 136        // max row width: 128 + 2*4 halo
#define RSTRIDE 36       // dwords per transposed row (64 ci = 32 dw, pad to 36: 16B-aligned, bank-spread)
#define WROWS 64
#define WSTRIDE 36       // dwords per weight row (64 ci = 32 dw, pad to 36)

// ---------------------------------------------------------------- pool mean
__global__ __launch_bounds__(256) void pool_mean_kernel(const float* __restrict__ x,
                                                        float* __restrict__ pooled) {
    __shared__ float red[256];
    const size_t base = (size_t)blockIdx.x * (HW * HW);
    float s = 0.f;
    for (int i = threadIdx.x; i < HW * HW; i += 256) s += x[base + i];
    red[threadIdx.x] = s;
    __syncthreads();
    for (int d = 128; d > 0; d >>= 1) {
        if (threadIdx.x < (unsigned)d) red[threadIdx.x] += red[threadIdx.x + d];
        __syncthreads();
    }
    if (threadIdx.x == 0) pooled[blockIdx.x] = red[0] * (1.f / (HW * HW));
}

// ---------------------------------------------------------------- zero fill
__global__ void zero_kernel(float* __restrict__ p, int n) {
    int i = blockIdx.x * blockDim.x + threadIdx.x;
    if (i < n) p[i] = 0.f;
}

// ------------------------------------------- attention + weight aggregation
// pooled: per-sample vector (stride cin), scale applied on read.
// w: [4][64][cin][3][3]; outputs hi/lo bf16 split in [16][9][64][cin].
__global__ __launch_bounds__(256) void attn_agg_kernel(
    const float* __restrict__ pooled, int cin, int hid, float scale,
    const float* __restrict__ fc1, const float* __restrict__ fc2w,
    const float* __restrict__ fc2b, const float* __restrict__ w,
    __hip_bfloat16* __restrict__ aggWhi, __hip_bfloat16* __restrict__ aggWlo) {
    const int b = blockIdx.x;
    __shared__ float sP[192];
    __shared__ float sH[64];
    __shared__ float sAtt[4];
    const int tid = threadIdx.x;

    if (tid < cin) sP[tid] = pooled[b * cin + tid] * scale;
    __syncthreads();
    if (tid < hid) {
        float h = 0.f;
        for (int c = 0; c < cin; ++c) h += fc1[tid * cin + c] * sP[c];
        sH[tid] = h > 0.f ? h : 0.f;
    }
    __syncthreads();
    if (tid == 0) {
        float l[4], e[4];
        float m = -1e30f;
        for (int k = 0; k < 4; ++k) {
            float s = fc2b[k];
            for (int j = 0; j < hid; ++j) s += sH[j] * fc2w[k * hid + j];
            l[k] = s * (1.f / 34.f);
            m = fmaxf(m, l[k]);
        }
        float den = 0.f;
        for (int k = 0; k < 4; ++k) { e[k] = __expf(l[k] - m); den += e[k]; }
        for (int k = 0; k < 4; ++k) sAtt[k] = e[k] / den;
    }
    __syncthreads();
    const float a0 = sAtt[0], a1 = sAtt[1], a2 = sAtt[2], a3 = sAtt[3];
    const int N = 9 * 64 * cin;
    const size_t kstride = (size_t)64 * cin * 9;
    for (int e = tid; e < N; e += 256) {
        const int tap = e / (64 * cin);
        const int rem = e - tap * (64 * cin);
        const int co = rem / cin;
        const int ci = rem - co * cin;
        const size_t si = ((size_t)(co * cin + ci)) * 9 + tap;
        const float v = a0 * w[si] + a1 * w[si + kstride] +
                        a2 * w[si + 2 * kstride] + a3 * w[si + 3 * kstride];
        const __hip_bfloat16 hi = __float2bfloat16(v);
        const __hip_bfloat16 lo = __float2bfloat16(v - __bfloat162float(hi));
        const size_t di = ((size_t)b * 9 + tap) * (64 * (size_t)cin) + (size_t)co * cin + ci;
        aggWhi[di] = hi;
        aggWlo[di] = lo;
    }
}

// ------------------------------------------------- implicit-GEMM dyn conv
// bf16x3 split GEMM on v_wmma_f32_16x16x32_bf16. One block = (y row, sample).
// M=64 Cout, N=128 pixels, K = 9 taps * inC. Input staged transposed [x][ci]
// so B's K direction is contiguous (two ds_load_b128 per operand).
__global__ __launch_bounds__(256) void dynconv_bf16x3_kernel(
    const float* __restrict__ inf32,            // fp32 input (split on the fly) or nullptr
    const __hip_bfloat16* __restrict__ inhi,    // pre-split input (trunk) ...
    const __hip_bfloat16* __restrict__ inlo,
    int inC, int nchunk, int dil,
    const __hip_bfloat16* __restrict__ whi,     // [16][9][64][inC]
    const __hip_bfloat16* __restrict__ wlo,
    __hip_bfloat16* __restrict__ cathi,         // split bf16 output (branches) or nullptr
    __hip_bfloat16* __restrict__ catlo,
    int outC, int chanOff,
    float* __restrict__ outf,                   // fp32 output (trunk) or nullptr
    const float* __restrict__ resid,            // residual (x) or nullptr
    float* __restrict__ pool, int poolStride, int poolOff,
    int leaky) {
    __shared__ unsigned int sWhi[WROWS * WSTRIDE];   //  9216 B
    __shared__ unsigned int sWlo[WROWS * WSTRIDE];   //  9216 B
    __shared__ unsigned int sBhi[BROWS * RSTRIDE];   // 19584 B
    __shared__ unsigned int sBlo[BROWS * RSTRIDE];   // 19584 B

    const int y = blockIdx.x;
    const int b = blockIdx.y;
    const int tid = threadIdx.x;
    const int lane = tid & 31;
    const int wid = tid >> 5;
    const int mtile = wid >> 1;            // 0..3 -> Cout tile
    const int ngrp = wid & 1;              // 0..1 -> N tile group
    const int nlo = lane & 15;
    const int h = lane >> 4;               // lane half (A/B K-half select)
    const int mhalf = h << 3;              // 0/8 (C/D row half)

    v8f acc[4];
#pragma unroll
    for (int t = 0; t < 4; ++t) acc[t] = (v8f){0, 0, 0, 0, 0, 0, 0, 0};

    const int rowW = HW + 2 * dil;
    const __hip_bfloat16 bz = __float2bfloat16(0.f);

    for (int chunk = 0; chunk < nchunk; ++chunk) {
        const int cibase = chunk * 64;
        for (int kh = 0; kh < 3; ++kh) {
            const int yy = y + (kh - 1) * dil;
            const bool rowValid = (yy >= 0) && (yy < HW);
            __syncthreads();  // protect previous tap's readers
            // ---- stage transposed split input row: [x][ci] bf16 ----
            if (inf32) {
                for (int e = tid; e < 64 * rowW; e += 256) {
                    const int ci = e / rowW;
                    const int xx = e - ci * rowW - dil;     // [-dil, HW+dil)
                    float v = 0.f;
                    if (rowValid && xx >= 0 && xx < HW)
                        v = inf32[(((size_t)b * inC + cibase + ci) * HW + yy) * HW + xx];
                    const __hip_bfloat16 vh = __float2bfloat16(v);
                    const __hip_bfloat16 vl = __float2bfloat16(v - __bfloat162float(vh));
                    const int xi = xx + dil;
                    ((__hip_bfloat16*)sBhi)[xi * (2 * RSTRIDE) + ci] = vh;
                    ((__hip_bfloat16*)sBlo)[xi * (2 * RSTRIDE) + ci] = vl;
                }
            } else {
                for (int e = tid; e < 64 * rowW; e += 256) {
                    const int ci = e / rowW;
                    const int xx = e - ci * rowW - dil;
                    __hip_bfloat16 vh = bz, vl = bz;
                    if (rowValid && xx >= 0 && xx < HW) {
                        const size_t gi = (((size_t)b * inC + cibase + ci) * HW + yy) * HW + xx;
                        vh = inhi[gi];
                        vl = inlo[gi];
                    }
                    const int xi = xx + dil;
                    ((__hip_bfloat16*)sBhi)[xi * (2 * RSTRIDE) + ci] = vh;
                    ((__hip_bfloat16*)sBlo)[xi * (2 * RSTRIDE) + ci] = vl;
                }
            }
            for (int kw = 0; kw < 3; ++kw) {
                const int tap = kh * 3 + kw;
                __syncthreads();
                // ---- stage hi/lo weight tile [co][ci] bf16 ----
                {
                    const size_t wb = ((size_t)b * 9 + tap) * (64 * (size_t)inC) + cibase;
                    const __hip_bfloat16* whsrc = whi + wb;
                    const __hip_bfloat16* wlsrc = wlo + wb;
                    for (int e = tid; e < 64 * 64; e += 256) {
                        const int co = e >> 6, ci = e & 63;
                        ((__hip_bfloat16*)sWhi)[co * (2 * WSTRIDE) + ci] = whsrc[(size_t)co * inC + ci];
                        ((__hip_bfloat16*)sWlo)[co * (2 * WSTRIDE) + ci] = wlsrc[(size_t)co * inC + ci];
                    }
                }
                __syncthreads();
                const int xsh = kw * dil;
#pragma unroll
                for (int k0d = 0; k0d < 32; k0d += 16) {   // K chunk of 32 (dword base 0/16)
                    // A (16x32 bf16): VGPR0-3 = K h*8+0..7, VGPR4-7 = K 16+h*8+0..7
                    union { v16bf v; uint4 q[2]; } Ah, Al;
                    const int arow = (mtile * 16 + nlo) * WSTRIDE + k0d + (h << 2);
                    Ah.q[0] = *(const uint4*)&sWhi[arow];
                    Ah.q[1] = *(const uint4*)&sWhi[arow + 8];
                    Al.q[0] = *(const uint4*)&sWlo[arow];
                    Al.q[1] = *(const uint4*)&sWlo[arow + 8];
#pragma unroll
                    for (int t = 0; t < 4; ++t) {
                        const int xi = (ngrp * 4 + t) * 16 + nlo + xsh;
                        // B (32x16 bf16): VGPR0-3 = K h*8+0..7, VGPR4-7 = K 16+h*8+0..7
                        union { v16bf v; uint4 q[2]; } Bh, Bl;
                        const int brow = xi * RSTRIDE + k0d + (h << 2);
                        Bh.q[0] = *(const uint4*)&sBhi[brow];
                        Bh.q[1] = *(const uint4*)&sBhi[brow + 8];
                        Bl.q[0] = *(const uint4*)&sBlo[brow];
                        Bl.q[1] = *(const uint4*)&sBlo[brow + 8];
                        acc[t] = __builtin_amdgcn_wmma_f32_16x16x32_bf16(
                            false, Ah.v, false, Bh.v, (short)0, acc[t], false, false);
                        acc[t] = __builtin_amdgcn_wmma_f32_16x16x32_bf16(
                            false, Ah.v, false, Bl.v, (short)0, acc[t], false, false);
                        acc[t] = __builtin_amdgcn_wmma_f32_16x16x32_bf16(
                            false, Al.v, false, Bh.v, (short)0, acc[t], false, false);
                    }
                }
            }
        }
    }

    // ---- epilogue ----
    if (leaky) {
#pragma unroll
        for (int t = 0; t < 4; ++t)
#pragma unroll
            for (int r = 0; r < 8; ++r) {
                const float v = acc[t][r];
                acc[t][r] = v >= 0.f ? v : 0.1f * v;
            }
    }
#pragma unroll
    for (int t = 0; t < 4; ++t) {
        const int xb = (ngrp * 4 + t) * 16;
#pragma unroll
        for (int r = 0; r < 8; ++r) {
            const int co = mtile * 16 + r + mhalf;
            const float v = acc[t][r];
            if (outf) {
                float o = v;
                if (resid)
                    o += resid[(((size_t)b * 64 + co) * HW + y) * HW + xb + nlo];
                outf[(((size_t)b * outC + chanOff + co) * HW + y) * HW + xb + nlo] = o;
            } else {
                const size_t oi = (((size_t)b * outC + chanOff + co) * HW + y) * HW + xb + nlo;
                const __hip_bfloat16 vh = __float2bfloat16(v);
                cathi[oi] = vh;
                catlo[oi] = __float2bfloat16(v - __bfloat162float(vh));
            }
        }
    }
    if (pool) {
#pragma unroll
        for (int r = 0; r < 8; ++r) {
            float s = acc[0][r] + acc[1][r] + acc[2][r] + acc[3][r];
            s += __shfl_xor(s, 1, 16);
            s += __shfl_xor(s, 2, 16);
            s += __shfl_xor(s, 4, 16);
            s += __shfl_xor(s, 8, 16);
            if (nlo == 0) {
                const int co = mtile * 16 + r + mhalf;
                atomicAdd(&pool[(size_t)b * poolStride + poolOff + co], s);
            }
        }
    }
}

// ---------------------------------------------------------------- launcher
extern "C" void kernel_launch(void* const* d_in, const int* in_sizes, int n_in,
                              void* d_out, int out_size, void* d_ws, size_t ws_size,
                              hipStream_t stream) {
    (void)in_sizes; (void)n_in; (void)out_size; (void)ws_size;
    const float* x = (const float*)d_in[0];
    const float* w_br[3]    = {(const float*)d_in[1], (const float*)d_in[5], (const float*)d_in[9]};
    const float* fc1_br[3]  = {(const float*)d_in[2], (const float*)d_in[6], (const float*)d_in[10]};
    const float* fc2w_br[3] = {(const float*)d_in[3], (const float*)d_in[7], (const float*)d_in[11]};
    const float* fc2b_br[3] = {(const float*)d_in[4], (const float*)d_in[8], (const float*)d_in[12]};
    const float* wt     = (const float*)d_in[13];
    const float* fc1_t  = (const float*)d_in[14];
    const float* fc2w_t = (const float*)d_in[15];
    const float* fc2b_t = (const float*)d_in[16];
    float* out = (float*)d_out;

    // ---- workspace layout (bytes) ----
    char* p = (char*)d_ws;
    float* pooled_x   = (float*)p;                  p += 4096;        // 16*64 f32
    float* pooled_cat = (float*)p;                  p += 16384;       // 16*192 f32
    const size_t aggB  = (size_t)16 * 9 * 64 * 64;  // per-branch elems (589824)
    const size_t aggT  = (size_t)16 * 9 * 64 * 192; // trunk elems (1769472)
    __hip_bfloat16* aggWhi = (__hip_bfloat16*)p;    p += 3 * aggB * 2;
    __hip_bfloat16* aggWlo = (__hip_bfloat16*)p;    p += 3 * aggB * 2;
    __hip_bfloat16* aggWthi = (__hip_bfloat16*)p;   p += aggT * 2;
    __hip_bfloat16* aggWtlo = (__hip_bfloat16*)p;   p += aggT * 2;
    const size_t catN = (size_t)16 * 192 * HW * HW; // 50331648
    __hip_bfloat16* cathi = (__hip_bfloat16*)p;     p += catN * 2;
    __hip_bfloat16* catlo = (__hip_bfloat16*)p;     p += catN * 2;

    const int dils[3] = {1, 2, 4};

    // 1) spatial mean of x -> pooled_x
    pool_mean_kernel<<<16 * 64, 256, 0, stream>>>(x, pooled_x);
    // 2) zero pooled_cat accumulators
    zero_kernel<<<(3072 + 255) / 256, 256, 0, stream>>>(pooled_cat, 3072);
    // 3) branch attention + split-weight aggregation (hid = 17)
    for (int br = 0; br < 3; ++br)
        attn_agg_kernel<<<16, 256, 0, stream>>>(pooled_x, 64, 17, 1.f,
                                                fc1_br[br], fc2w_br[br], fc2b_br[br],
                                                w_br[br], aggWhi + br * aggB, aggWlo + br * aggB);
    // 4) branch convs (fp32 input split on the fly) -> split bf16 cat (+pool atomics)
    for (int br = 0; br < 3; ++br)
        dynconv_bf16x3_kernel<<<dim3(HW, 16), 256, 0, stream>>>(
            x, nullptr, nullptr, 64, 1, dils[br],
            aggWhi + br * aggB, aggWlo + br * aggB,
            cathi, catlo, 192, br * 64,
            nullptr, nullptr, pooled_cat, 192, br * 64, 1);
    // 5) trunk attention + aggregation (cin=192, hid=49, mean scale)
    attn_agg_kernel<<<16, 256, 0, stream>>>(pooled_cat, 192, 49, 1.f / (HW * HW),
                                            fc1_t, fc2w_t, fc2b_t, wt, aggWthi, aggWtlo);
    // 6) trunk conv (bf16 cat) + fp32 residual -> d_out
    dynconv_bf16x3_kernel<<<dim3(HW, 16), 256, 0, stream>>>(
        nullptr, cathi, catlo, 192, 3, 1,
        aggWthi, aggWtlo,
        nullptr, nullptr, 64, 0,
        out, x, nullptr, 0, 0, 0);
}
